// Extractor_26860725469209
// MI455X (gfx1250) — matrix operand order repaired and checked by reading, compiled, not run on hardware
//
#include <hip/hip_runtime.h>

// ---------------------------------------------------------------------------
// Problem constants (from reference)
// ---------------------------------------------------------------------------
#define PNUM   256      // polynomials
#define MNUM   512      // monomials per polynomial
#define NVAR   16
#define DM     128      // embedding dim (both stages)
#define NLAYER 4
#define NROWS  (PNUM * MNUM)        // 131072 monomial rows
#define CHROWS 32768                // rows per streamed chunk (64 polys)
#define NCHUNK (NROWS / CHROWS)     // 4
#define EPS    1e-5f

// LDS tile strides (padded to dodge bank conflicts)
#define FSTR 132   // fp32 tile row stride (floats)
#define HSTR 136   // bf16 tile row stride (halfs)

typedef __bf16 v16bf __attribute__((ext_vector_type(16)));
typedef float  v8f   __attribute__((ext_vector_type(8)));

// ---------------------------------------------------------------------------
// Wave-level 16x128 GEMM over K=128 using V_WMMA_F32_16X16X32_BF16.
//   th   : this wave's LDS bf16 tile [16][HSTR]  (A rows)
//   blob : weight fragments, pre-swizzled: scalar index
//          ((ntile*4 + ktile)*32 + lane)*16 + e,  where
//          B[k][c] element lives at k = ktile*32 + (lane>>4)*16 + e,
//                                   c = ntile*16 + (lane&15)
//   acc  : 8 accumulators = full 16x128 fp32 tile (C/D layout per ISA 7.12.2)
// A fragment layout used here (16x32, contiguous k, same order as blob B):
//   lane: m=lane&15, half=lane>>4; element e:
//   kloc = half<<4 | e
// ---------------------------------------------------------------------------
__device__ __forceinline__ void wave_gemm128(const __bf16* th, const __bf16* blob,
                                             int lane, v8f acc[8]) {
  const v16bf* frag = (const v16bf*)blob;
  const int m = lane & 15, half = lane >> 4;
#pragma unroll
  for (int kt = 0; kt < 4; ++kt) {
    v16bf a;
#pragma unroll
    for (int e = 0; e < 16; ++e) {
      const int kloc = (half << 4) + e;
      a[e] = th[m * HSTR + kt * 32 + kloc];
    }
#pragma unroll
    for (int n = 0; n < 8; ++n) {
      v16bf b = frag[(n * 4 + kt) * 32 + lane];
      acc[n] = __builtin_amdgcn_wmma_f32_16x16x32_bf16(
          false, a, false, b, (short)0, acc[n], false, false);
    }
  }
}

// ---------------------------------------------------------------------------
// Wave-level row LayerNorm over the fp32 LDS tile, then coalesced store.
// tf already holds (residual + gemm + bias). reds/redq: per-wave [32] scratch.
// ---------------------------------------------------------------------------
__device__ __forceinline__ void wave_ln_store(float* tf, float* reds, float* redq,
                                              const float* gamma, const float* beta,
                                              float* out, int rowbase, int lane) {
  const int rowl = lane & 15;
  const int hh = (lane >> 4) * 64;
  float s = 0.f, q2 = 0.f;
#pragma unroll 8
  for (int c = 0; c < 64; ++c) {
    float v = tf[rowl * FSTR + hh + c];
    s += v; q2 += v * v;
  }
  reds[lane] = s; redq[lane] = q2;
  __syncthreads();
  const float mu = (reds[lane] + reds[lane ^ 16]) * (1.f / 128.f);
  const float m2 = (redq[lane] + redq[lane ^ 16]) * (1.f / 128.f);
  const float rstd = rsqrtf(m2 - mu * mu + EPS);
#pragma unroll 8
  for (int c = 0; c < 64; ++c) {
    const int cc = hh + c;
    tf[rowl * FSTR + cc] = (tf[rowl * FSTR + cc] - mu) * rstd * gamma[cc] + beta[cc];
  }
  __syncthreads();
#pragma unroll 4
  for (int i = 0; i < 64; ++i) {
    const int idx = i * 32 + lane, row = idx >> 7, col = idx & 127;
    out[(rowbase + row) * 128 + col] = tf[row * FSTR + col];
  }
}

// ---------------------------------------------------------------------------
// GEMM kernel: out = g @ B + bias  (mode 0: plain fp32 store)
//              out = LN(r + g @ B + bias) (mode 1: residual + LayerNorm)
// 4 waves / block, each wave owns a 16-row tile. Rows per block = 64.
// In-place safe (row i output depends only on row i inputs).
// ---------------------------------------------------------------------------
__global__ void __launch_bounds__(128)
gemm_ln_kernel(const float* __restrict__ g, const float* __restrict__ r,
               float* __restrict__ out, const __bf16* __restrict__ wblob,
               const float* __restrict__ bias, const float* __restrict__ gamma,
               const float* __restrict__ beta, int mode) {
  __shared__ float  tf[4][16][FSTR];
  __shared__ __bf16 th[4][16][HSTR];
  __shared__ float  reds[4][32], redq[4][32];
  const int wave = threadIdx.x >> 5, lane = threadIdx.x & 31;
  const int rowbase = (blockIdx.x * 4 + wave) * 16;
  float*  tfw = &tf[wave][0][0];
  __bf16* thw = &th[wave][0][0];

#pragma unroll 4
  for (int i = 0; i < 64; ++i) {
    const int idx = i * 32 + lane, row = idx >> 7, col = idx & 127;
    const float gv = g[(rowbase + row) * 128 + col];
    thw[row * HSTR + col] = (__bf16)gv;
    if (mode) tfw[row * FSTR + col] = r[(rowbase + row) * 128 + col];
  }
  __syncthreads();

  v8f acc[8];
#pragma unroll
  for (int n = 0; n < 8; ++n) {
#pragma unroll
    for (int e = 0; e < 8; ++e) acc[n][e] = 0.f;
  }
  wave_gemm128(thw, wblob, lane, acc);

  const int cm = lane & 15, half = lane >> 4;
  if (mode == 0) {
#pragma unroll
    for (int n = 0; n < 8; ++n) {
      const float bv = bias[n * 16 + cm];
#pragma unroll
      for (int rr = 0; rr < 8; ++rr)
        out[(rowbase + rr + (half << 3)) * 128 + n * 16 + cm] = acc[n][rr] + bv;
    }
    return;
  }
#pragma unroll
  for (int n = 0; n < 8; ++n) {
    const float bv = bias[n * 16 + cm];
#pragma unroll
    for (int rr = 0; rr < 8; ++rr)
      tfw[(rr + (half << 3)) * FSTR + n * 16 + cm] += acc[n][rr] + bv;
  }
  __syncthreads();
  wave_ln_store(tfw, &reds[wave][0], &redq[wave][0], gamma, beta, out, rowbase, lane);
}

// ---------------------------------------------------------------------------
// Fused FFN: x = LN(x + relu(x@W1+b1)@W2 + b2). Hidden tile never leaves LDS.
// ---------------------------------------------------------------------------
__global__ void __launch_bounds__(128)
ffn_kernel(float* __restrict__ x, const __bf16* __restrict__ w1b,
           const float* __restrict__ b1, const __bf16* __restrict__ w2b,
           const float* __restrict__ b2, const float* __restrict__ gamma,
           const float* __restrict__ beta) {
  __shared__ float  tf[4][16][FSTR];
  __shared__ __bf16 th[4][16][HSTR];
  __shared__ float  reds[4][32], redq[4][32];
  const int wave = threadIdx.x >> 5, lane = threadIdx.x & 31;
  const int rowbase = (blockIdx.x * 4 + wave) * 16;
  float*  tfw = &tf[wave][0][0];
  __bf16* thw = &th[wave][0][0];

#pragma unroll 4
  for (int i = 0; i < 64; ++i) {
    const int idx = i * 32 + lane, row = idx >> 7, col = idx & 127;
    const float v = x[(rowbase + row) * 128 + col];
    tfw[row * FSTR + col] = v;
    thw[row * HSTR + col] = (__bf16)v;
  }
  __syncthreads();

  v8f acc[8];
#pragma unroll
  for (int n = 0; n < 8; ++n) {
#pragma unroll
    for (int e = 0; e < 8; ++e) acc[n][e] = 0.f;
  }
  wave_gemm128(thw, w1b, lane, acc);
  __syncthreads();

  const int cm = lane & 15, half = lane >> 4;
#pragma unroll
  for (int n = 0; n < 8; ++n) {
    const float bv = b1[n * 16 + cm];
#pragma unroll
    for (int rr = 0; rr < 8; ++rr)
      thw[(rr + (half << 3)) * HSTR + n * 16 + cm] =
          (__bf16)fmaxf(acc[n][rr] + bv, 0.f);
  }
  __syncthreads();

#pragma unroll
  for (int n = 0; n < 8; ++n) {
#pragma unroll
    for (int e = 0; e < 8; ++e) acc[n][e] = 0.f;
  }
  wave_gemm128(thw, w2b, lane, acc);
#pragma unroll
  for (int n = 0; n < 8; ++n) {
    const float bv = b2[n * 16 + cm];
#pragma unroll
    for (int rr = 0; rr < 8; ++rr)
      tfw[(rr + (half << 3)) * FSTR + n * 16 + cm] += acc[n][rr] + bv;
  }
  __syncthreads();
  wave_ln_store(tfw, &reds[wave][0], &redq[wave][0], gamma, beta, x, rowbase, lane);
}

// ---------------------------------------------------------------------------
// Weight packing: B[k][c] = W[c][k] into WMMA fragment order (bf16).
// ---------------------------------------------------------------------------
__global__ void __launch_bounds__(256)
pack_plain_kernel(const float* __restrict__ W, __bf16* __restrict__ frag) {
  const int t = blockIdx.x * 256 + threadIdx.x;      // 0..16383
  const int e = t & 15, lane = (t >> 4) & 31, kt = (t >> 9) & 3, nt = t >> 11;
  const int k = kt * 32 + ((lane >> 4) << 4) + e;
  const int c = nt * 16 + (lane & 15);
  frag[t] = (__bf16)W[c * 128 + k];
}

// Fused seq-len-1 attention weight: B[k][c] = sum_j Wo[c][j] * Wv[j][k]
__global__ void __launch_bounds__(256)
pack_weff_kernel(const float* __restrict__ wqkv, const float* __restrict__ wo,
                 __bf16* __restrict__ frag) {
  const int t = blockIdx.x * 256 + threadIdx.x;
  const int e = t & 15, lane = (t >> 4) & 31, kt = (t >> 9) & 3, nt = t >> 11;
  const int k = kt * 32 + ((lane >> 4) << 4) + e;
  const int c = nt * 16 + (lane & 15);
  const float* wv = wqkv + 256 * 128;                // V block of in_proj
  float s = 0.f;
  for (int j = 0; j < 128; ++j) s += wo[c * 128 + j] * wv[j * 128 + k];
  frag[t] = (__bf16)s;
}

// beff[c] = bo[c] + sum_j Wo[c][j] * bv[j]
__global__ void __launch_bounds__(128)
beff_kernel(const float* __restrict__ wo, const float* __restrict__ bv,
            const float* __restrict__ bo, float* __restrict__ beff) {
  const int c = threadIdx.x;
  float s = bo[c];
  for (int j = 0; j < 128; ++j) s += wo[c * 128 + j] * bv[j];
  beff[c] = s;
}

// ---------------------------------------------------------------------------
// Monomial embedding for one chunk: x[row] = mono[row] @ mono_w^T + mono_b
// ---------------------------------------------------------------------------
__global__ void __launch_bounds__(256)
embed_kernel(const float* __restrict__ mono, const float* __restrict__ w,
             const float* __restrict__ b, float* __restrict__ out) {
  const int base = blockIdx.x * 16384;               // 128 rows per block
  for (int i = 0; i < 64; ++i) {
    const int idx = base + i * 256 + threadIdx.x;
    const int row = idx >> 7, col = idx & 127;
    const float* mr = mono + row * NVAR;
    const float* wr = w + col * NVAR;
    float acc = b[col];
#pragma unroll
    for (int k = 0; k < NVAR; ++k) acc += mr[k] * wr[k];
    out[idx] = acc;
  }
}

// Mean over MNUM monomials for the 64 polys in this chunk.
__global__ void __launch_bounds__(128)
mean_kernel(const float* __restrict__ xch, float* __restrict__ pf, int poly_base) {
  const int p = blockIdx.x, c = threadIdx.x;
  float s = 0.f;
  for (int m = 0; m < MNUM; ++m) s += xch[(p * MNUM + m) * 128 + c];
  pf[(poly_base + p) * 128 + c] = s * (1.f / (float)MNUM);
}

// ---------------------------------------------------------------------------
// Ideal-stage attention: one block per (head, query). dh=16, S=256, scale=1/4.
// ---------------------------------------------------------------------------
__global__ void __launch_bounds__(64)
attn_kernel(const float* __restrict__ q, const float* __restrict__ k,
            const float* __restrict__ v, float* __restrict__ o) {
  __shared__ float s[256], qs[16], red[64], bmax, bsum;
  const int h = blockIdx.x >> 8, qi = blockIdx.x & 255, t = threadIdx.x;
  if (t < 16) qs[t] = q[qi * 128 + h * 16 + t];
  __syncthreads();
  float lm = -1e30f;
  for (int j = t; j < 256; j += 64) {
    float d = 0.f;
#pragma unroll
    for (int e = 0; e < 16; ++e) d += qs[e] * k[j * 128 + h * 16 + e];
    d *= 0.25f;
    s[j] = d;
    lm = fmaxf(lm, d);
  }
  red[t] = lm; __syncthreads();
  if (t == 0) { float m = red[0]; for (int i = 1; i < 64; ++i) m = fmaxf(m, red[i]); bmax = m; }
  __syncthreads();
  float ls = 0.f;
  for (int j = t; j < 256; j += 64) { float e = expf(s[j] - bmax); s[j] = e; ls += e; }
  red[t] = ls; __syncthreads();
  if (t == 0) { float ss = 0.f; for (int i = 0; i < 64; ++i) ss += red[i]; bsum = ss; }
  __syncthreads();
  if (t < 16) {
    float acc = 0.f;
    for (int j = 0; j < 256; ++j) acc += s[j] * v[j * 128 + h * 16 + t];
    o[qi * 128 + h * 16 + t] = acc / bsum;
  }
}

// Final pair values: out[i][j] = dot(y[i], y[j])
__global__ void __launch_bounds__(256)
outer_kernel(const float* __restrict__ y, float* __restrict__ out) {
  const int i = blockIdx.x, j = threadIdx.x;
  float s = 0.f;
#pragma unroll 8
  for (int c = 0; c < 128; ++c) s += y[i * 128 + c] * y[j * 128 + c];
  out[i * 256 + j] = s;
}

// ---------------------------------------------------------------------------
// Host-side orchestration
// ---------------------------------------------------------------------------
extern "C" void kernel_launch(void* const* d_in, const int* in_sizes, int n_in,
                              void* d_out, int out_size, void* d_ws, size_t ws_size,
                              hipStream_t stream) {
  (void)n_in; (void)out_size; (void)ws_size;
  const float* monomials = (const float*)d_in[0];
  const float* mono_w    = (const float*)d_in[1];
  const float* mono_b    = (const float*)d_in[2];

  // Layer-dict flattening order: detect insertion-order vs sorted-keys.
  // insertion: wqkv(196608),bqkv,wo,bo,w1,b1,w2,b2,ln1_s,ln1_b,ln2_s,ln2_b
  // sorted   : b1(512),b2,bo,bqkv,ln1_b,ln1_s,ln2_b,ln2_s,w1,w2,wo,wqkv
  const bool sorted_keys = (in_sizes[3] != 4 * 3 * 128 * 128);
  // map[i] = position of {wqkv,bqkv,wo,bo,w1,b1,w2,b2,ln1_s,ln1_b,ln2_s,ln2_b}
  static const int map_ins[12] = {0, 1, 2, 3, 4, 5, 6, 7, 8, 9, 10, 11};
  static const int map_srt[12] = {11, 3, 10, 2, 8, 0, 9, 1, 5, 4, 7, 6};
  const int* mp = sorted_keys ? map_srt : map_ins;

  const float* PL[12]; const float* IL[12];
  for (int i = 0; i < 12; ++i) PL[i] = (const float*)d_in[3 + mp[i]];
  const float* pemb_w = (const float*)d_in[15];
  const float* pemb_b = (const float*)d_in[16];
  for (int i = 0; i < 12; ++i) IL[i] = (const float*)d_in[17 + mp[i]];

  enum { WQKV, BQKV, WO, BO, W1, B1, W2, B2, LN1S, LN1B, LN2S, LN2B };

  // Workspace layout
  char* wsb = (char*)d_ws;
  float*  xch   = (float*)(wsb + 0);                       // 32768*128 f32 = 16 MB
  __bf16* frags = (__bf16*)(wsb + 16777216);               // 37 blobs * 16384 bf16
  float*  beff  = (float*)(wsb + 17989632);                // 4*128
  float*  pf    = (float*)(wsb + 17991680);                // 256*128
  float*  xi    = (float*)(wsb + 18122752);                // 256*128
  float*  qb    = (float*)(wsb + 18253824);
  float*  kb    = (float*)(wsb + 18384896);
  float*  vb    = (float*)(wsb + 18515968);
  float*  ao    = (float*)(wsb + 18647040);
  auto blob = [&](int i) { return frags + (size_t)i * 16384; };
  // blob indices: poly l -> {weff: 3l, w1: 3l+1, w2: 3l+2}; pemb: 12;
  //               ideal l -> 13+6l + {q,k,v,o,w1,w2}

  // ---- prep: pack all weights into WMMA fragment blobs ----
  for (int l = 0; l < NLAYER; ++l) {
    pack_weff_kernel<<<64, 256, 0, stream>>>(PL[WQKV] + l * 384 * 128,
                                             PL[WO] + l * 128 * 128, blob(3 * l));
    beff_kernel<<<1, 128, 0, stream>>>(PL[WO] + l * 128 * 128,
                                       PL[BQKV] + l * 384 + 256,
                                       PL[BO] + l * 128, beff + l * 128);
    pack_plain_kernel<<<64, 256, 0, stream>>>(PL[W1] + l * 16384, blob(3 * l + 1));
    pack_plain_kernel<<<64, 256, 0, stream>>>(PL[W2] + l * 16384, blob(3 * l + 2));
  }
  pack_plain_kernel<<<64, 256, 0, stream>>>(pemb_w, blob(12));
  for (int l = 0; l < NLAYER; ++l) {
    const float* wq = IL[WQKV] + l * 384 * 128;
    pack_plain_kernel<<<64, 256, 0, stream>>>(wq, blob(13 + 6 * l + 0));
    pack_plain_kernel<<<64, 256, 0, stream>>>(wq + 16384, blob(13 + 6 * l + 1));
    pack_plain_kernel<<<64, 256, 0, stream>>>(wq + 32768, blob(13 + 6 * l + 2));
    pack_plain_kernel<<<64, 256, 0, stream>>>(IL[WO] + l * 16384, blob(13 + 6 * l + 3));
    pack_plain_kernel<<<64, 256, 0, stream>>>(IL[W1] + l * 16384, blob(13 + 6 * l + 4));
    pack_plain_kernel<<<64, 256, 0, stream>>>(IL[W2] + l * 16384, blob(13 + 6 * l + 5));
  }

  // ---- poly stage: stream 4 L2-resident chunks of 32768 rows ----
  for (int ch = 0; ch < NCHUNK; ++ch) {
    embed_kernel<<<CHROWS / 128, 256, 0, stream>>>(
        monomials + (size_t)ch * CHROWS * NVAR, mono_w, mono_b, xch);
    for (int l = 0; l < NLAYER; ++l) {
      // x = LN(x + x @ (Wo Wv)^T + beff)   [seq-len-1 attention, fused]
      gemm_ln_kernel<<<CHROWS / 64, 128, 0, stream>>>(
          xch, xch, xch, blob(3 * l), beff + l * 128,
          PL[LN1S] + l * 128, PL[LN1B] + l * 128, 1);
      // x = LN(x + relu(x W1 + b1) W2 + b2)
      ffn_kernel<<<CHROWS / 64, 128, 0, stream>>>(
          xch, blob(3 * l + 1), PL[B1] + l * 128, blob(3 * l + 2),
          PL[B2] + l * 128, PL[LN2S] + l * 128, PL[LN2B] + l * 128);
    }
    mean_kernel<<<64, 128, 0, stream>>>(xch, pf, ch * 64);
  }

  // ---- poly embedding projection: xi = pf @ pemb^T + b ----
  gemm_ln_kernel<<<4, 128, 0, stream>>>(pf, pf, xi, blob(12), pemb_b,
                                        nullptr, nullptr, 0);

  // ---- ideal stage: real attention over 256 polys ----
  for (int l = 0; l < NLAYER; ++l) {
    const __bf16* fq = blob(13 + 6 * l + 0);
    const __bf16* fk = blob(13 + 6 * l + 1);
    const __bf16* fv = blob(13 + 6 * l + 2);
    gemm_ln_kernel<<<4, 128, 0, stream>>>(xi, xi, qb, fq, IL[BQKV] + l * 384 + 0,
                                          nullptr, nullptr, 0);
    gemm_ln_kernel<<<4, 128, 0, stream>>>(xi, xi, kb, fk, IL[BQKV] + l * 384 + 128,
                                          nullptr, nullptr, 0);
    gemm_ln_kernel<<<4, 128, 0, stream>>>(xi, xi, vb, fv, IL[BQKV] + l * 384 + 256,
                                          nullptr, nullptr, 0);
    attn_kernel<<<8 * 256, 64, 0, stream>>>(qb, kb, vb, ao);
    gemm_ln_kernel<<<4, 128, 0, stream>>>(ao, xi, xi, blob(13 + 6 * l + 3),
                                          IL[BO] + l * 128, IL[LN1S] + l * 128,
                                          IL[LN1B] + l * 128, 1);
    ffn_kernel<<<4, 128, 0, stream>>>(xi, blob(13 + 6 * l + 4), IL[B1] + l * 128,
                                      blob(13 + 6 * l + 5), IL[B2] + l * 128,
                                      IL[LN2S] + l * 128, IL[LN2B] + l * 128);
  }

  // ---- pair values: out = y @ y^T ----
  outer_kernel<<<256, 256, 0, stream>>>(xi, (float*)d_out);
}
